// ShiftedWindowMSA_10651518894306
// MI455X (gfx1250) — matrix operand order repaired and checked by reading, not compiled, not run on hardware
//
#include <hip/hip_runtime.h>
#include <hip/hip_bf16.h>

// ---------------------------------------------------------------------------
// Shifted-window MSA with Tucker-factorized QKV, mapped to CDNA5 WMMA.
//  - prep kernel: densify W1(x)W2(x)W3 -> 3x [384x384] bf16 with rows permuted
//    into (head, head_channel) order; permuted biases.
//  - main kernel: 1 workgroup per window (2048 windows), 4 wave32,
//    288KB LDS, v_wmma_f32_16x16x32_bf16 for projections / QK^T / A^T V.
// ---------------------------------------------------------------------------

typedef __bf16 bf16;
typedef __bf16 v16bf __attribute__((ext_vector_type(16)));
typedef float  v8f   __attribute__((ext_vector_type(8)));

#define EMB   384
#define TOK   49
#define TPAD  64
#define HDIM  32
#define SHIFT 3
#define IMGD  56
#define SCALE_C 0.17677669529663687f   // 32^-0.5

union Frag16 { v16bf v; unsigned u[8]; };

__device__ __forceinline__ v8f vzero() {
  v8f z = {0.f,0.f,0.f,0.f,0.f,0.f,0.f,0.f};
  return z;
}

__device__ __forceinline__ unsigned pack2(float a, float b) {
  unsigned short ux = __builtin_bit_cast(unsigned short, (bf16)a);
  unsigned short uy = __builtin_bit_cast(unsigned short, (bf16)b);
  return (unsigned)ux | ((unsigned)uy << 16);
}

// A operand, 16x32 bf16: row M = lane&15; lanes<16: K=0..15,
// lanes>=16: K=16..31 (contiguous).  K-pairs are contiguous -> 32-bit loads.
__device__ __forceinline__ v16bf load_a(const bf16* base, int row0, int k0,
                                        int stride, int lane) {
  Frag16 f;
  const unsigned* p = (const unsigned*)(base + (row0 + (lane & 15)) * stride);
  int g8 = (lane >> 4) << 3;
#pragma unroll
  for (int i = 0; i < 8; ++i) {
    int kk = 2 * i + 2 * g8 + k0;  // even
    f.u[i] = p[kk >> 1];
  }
  return f.v;
}

// B operand, 32x16 bf16: element[k][col]; lanes<16: col=lane, K=2p,2p+1;
// lanes>=16: col=lane-16, K=16+2p.  Source matrix is [col][k] row-major so
// K-pairs are contiguous -> 32-bit loads (works for LDS and global).
__device__ __forceinline__ v16bf load_b(const bf16* base, int col0, int k0,
                                        int stride, int lane) {
  Frag16 f;
  const unsigned* p = (const unsigned*)(base + (col0 + (lane & 15)) * stride);
  int kg = ((lane >> 4) << 4) + k0;
#pragma unroll
  for (int i = 0; i < 8; ++i) f.u[i] = p[(kg + 2 * i) >> 1];
  return f.v;
}

__device__ __forceinline__ v8f wmma_bf16(v16bf a, v16bf b, v8f c) {
  return __builtin_amdgcn_wmma_f32_16x16x32_bf16(false, a, false, b,
                                                 (short)0, c, false, false);
}

__device__ __forceinline__ int catf(int r) { return (r >= 53) ? 2 : ((r >= 49) ? 1 : 0); }

__device__ __forceinline__ float maskval(int m, int n, int t, int s) {
  int lt = catf(m * 7 + t / 7) * 3 + catf(n * 7 + t % 7);
  int ls = catf(m * 7 + s / 7) * 3 + catf(n * 7 + s % 7);
  return (lt == ls) ? 0.f : -100.f;
}

// ---------------------------------------------------------------------------
// Prep: dense weights (row = h*32+c permuted) + permuted biases.
// ---------------------------------------------------------------------------
__global__ void __launch_bounds__(EMB) swmsa_prep(
    const float* __restrict__ Wq1, const float* __restrict__ Wq2,
    const float* __restrict__ Wq3, const float* __restrict__ bq,
    const float* __restrict__ Wk1, const float* __restrict__ Wk2,
    const float* __restrict__ Wk3, const float* __restrict__ bk,
    const float* __restrict__ Wv1, const float* __restrict__ Wv2,
    const float* __restrict__ Wv3, const float* __restrict__ bv,
    bf16* __restrict__ Wd, float* __restrict__ biasPerm) {
  int row = blockIdx.x % EMB;
  int mat = blockIdx.x / EMB;
  const float* W1 = (mat == 0) ? Wq1 : ((mat == 1) ? Wk1 : Wv1);
  const float* W2 = (mat == 0) ? Wq2 : ((mat == 1) ? Wk2 : Wv2);
  const float* W3 = (mat == 0) ? Wq3 : ((mat == 1) ? Wk3 : Wv3);
  const float* bb = (mat == 0) ? bq  : ((mat == 1) ? bk  : bv);
  // row -> head h=(a*2+b)*3+g ; c=(u*4+v)*2+w ; input-side split: p=u*2+a ...
  int h = row >> 5, c = row & 31;
  int gam = h % 3, bet = (h / 3) % 2, alp = h / 6;
  int w2 = c & 1, v4 = (c >> 1) & 3, u4 = c >> 2;
  int p = u4 * 2 + alp, q = v4 * 2 + bet, s3 = w2 * 3 + gam;
  int col = threadIdx.x;                    // input (e,f,g) flat
  int e = col / 48, f = (col / 6) % 8, g = col % 6;
  float val = W1[p * 8 + e] * W2[q * 8 + f] * W3[s3 * 6 + g];
  Wd[(size_t)mat * EMB * EMB + (size_t)row * EMB + col] = (bf16)val;
  if (col == 0) biasPerm[mat * EMB + row] = bb[(p * 8 + q) * 6 + s3];
}

// ---------------------------------------------------------------------------
// Main: one workgroup per window.
// LDS: Xb[64][384] Qb[64][384] Kb[64][384] Vt[384][64] (bf16)
//      Sw[4][64][64] f32, At[4][64][64] bf16   => 294912 bytes
// ---------------------------------------------------------------------------
__global__ void __launch_bounds__(128) swmsa_main(
    const float* __restrict__ x, const bf16* __restrict__ Wd,
    const float* __restrict__ biasPerm, float* __restrict__ out) {
  extern __shared__ __align__(16) char smem[];
  bf16*  Xb  = (bf16*)smem;                 // 49152 B
  bf16*  Qb  = Xb + TPAD * EMB;             // 49152 B
  bf16*  Kb  = Qb + TPAD * EMB;             // 49152 B
  bf16*  Vt  = Kb + TPAD * EMB;             // 49152 B  [chan][tok]
  float* Sw  = (float*)(Vt + EMB * TPAD);   // 65536 B
  bf16*  Atw = (bf16*)(Sw + 4 * 64 * 64);   // 32768 B  [s][t] transposed

  const int tid  = threadIdx.x;
  const int lane = tid & 31;
  const int wv   = tid >> 5;
  const int blk  = blockIdx.x;
  const int n = blk & 7, m = (blk >> 3) & 7, b = blk >> 6;
  const int g8 = (lane >> 4) << 3;

  // ---- 1) gather shifted window, fp32 -> bf16 into LDS -------------------
  for (int t = 0; t < TOK; ++t) {
    int hsrc = m * 7 + (t / 7) + SHIFT; if (hsrc >= IMGD) hsrc -= IMGD;
    int wsrc = n * 7 + (t % 7) + SHIFT; if (wsrc >= IMGD) wsrc -= IMGD;
    const float4* src =
        (const float4*)(x + (((size_t)b * IMGD + hsrc) * IMGD + wsrc) * EMB);
    for (int c4 = tid; c4 < EMB / 4; c4 += 128) {
      float4 vv = src[c4];
      uint2 pk; pk.x = pack2(vv.x, vv.y); pk.y = pack2(vv.z, vv.w);
      *(uint2*)(Xb + t * EMB + c4 * 4) = pk;
    }
  }
  for (int i = tid; i < (TPAD - TOK) * EMB / 2; i += 128)
    ((unsigned*)(Xb + TOK * EMB))[i] = 0u;   // zero pad rows 49..63
  __syncthreads();

  // ---- 2) projections: Q/K/V = X * Wd^T + bias (WMMA bf16) ---------------
  for (int wc = wv; wc < 3 * 24; wc += 4) {  // (matrix, 16-col tile)
    int mat = wc / 24, ct = wc % 24;
    const bf16* W = Wd + (size_t)mat * EMB * EMB;
    v8f acc[4];
#pragma unroll
    for (int r = 0; r < 4; ++r) acc[r] = vzero();
    for (int kb = 0; kb < EMB / 32; ++kb) {
      v16bf bfr = load_b(W, ct * 16, kb * 32, EMB, lane);  // weights via L2
#pragma unroll
      for (int rt = 0; rt < 4; ++rt) {
        v16bf af = load_a(Xb, rt * 16, kb * 32, EMB, lane);
        acc[rt] = wmma_bf16(af, bfr, acc[rt]);
      }
    }
    int   chan = ct * 16 + (lane & 15);
    float bia  = biasPerm[mat * EMB + chan];
#pragma unroll
    for (int rt = 0; rt < 4; ++rt) {
#pragma unroll
      for (int r = 0; r < 8; ++r) {
        int   tok = rt * 16 + r + g8;                 // D: M = r + 8*(lane>>4)
        float val = (tok < TOK) ? (acc[rt][r] + bia) : 0.f;
        if (mat == 0)      Qb[tok * EMB + chan]  = (bf16)val;
        else if (mat == 1) Kb[tok * EMB + chan]  = (bf16)val;
        else               Vt[chan * TPAD + tok] = (bf16)val;  // transposed
      }
    }
  }
  __syncthreads();

  // ---- 3) attention: each wave owns 3 heads ------------------------------
  float* S  = Sw  + wv * 64 * 64;
  bf16*  At = Atw + wv * 64 * 64;
  for (int hh = 0; hh < 3; ++hh) {
    int h = wv * 3 + hh;
    int hc = h * HDIM;

    // S = (Q Kh^T)*scale + shift-mask
    for (int tt = 0; tt < 4; ++tt) {
      v16bf af = load_a(Qb, tt * 16, hc, EMB, lane);
      for (int ts = 0; ts < 4; ++ts) {
        v16bf bfr = load_b(Kb, ts * 16, hc, EMB, lane);
        v8f c = wmma_bf16(af, bfr, vzero());
        int scol = ts * 16 + (lane & 15);
#pragma unroll
        for (int r = 0; r < 8; ++r) {
          int   trow = tt * 16 + r + g8;
          float val  = 0.f;
          if (trow < TOK && scol < TOK)
            val = c[r] * SCALE_C + maskval(m, n, trow, scol);
          S[trow * 64 + scol] = val;
        }
      }
    }
    asm volatile("s_wait_dscnt 0x0" ::: "memory");

    // signed softmax over s; store A^T (bf16) for the A^T*V GEMM
#pragma unroll
    for (int pp = 0; pp < 2; ++pp) {
      int t = lane + pp * 32;
      float mx = 0.f;
      for (int s = 0; s < TOK; ++s) mx = fmaxf(mx, fabsf(S[t * 64 + s]));
      float sum = 0.f;
      for (int s = 0; s < TOK; ++s) sum += __expf(fabsf(S[t * 64 + s]) - mx);
      float inv = 1.f / sum;
      for (int s = 0; s < 64; ++s) {
        float r = 0.f;
        if (s < TOK) {
          float v = S[t * 64 + s];
          float e = __expf(fabsf(v) - mx) * inv;
          r = (v > 0.f) ? e : ((v < 0.f) ? -e : 0.f);
        }
        At[s * 64 + t] = (bf16)r;
      }
    }
    asm volatile("s_wait_dscnt 0x0" ::: "memory");

    // O[s, c] = sum_t A[t,s] V[t,c]  (out = A^T V), store with output perm
    for (int st = 0; st < 4; ++st) {
      for (int ctile = 0; ctile < 2; ++ctile) {
        v8f c = vzero();
#pragma unroll
        for (int kt = 0; kt < 2; ++kt) {
          v16bf af  = load_a(At, st * 16, kt * 32, 64, lane);
          v16bf bfr = load_b(Vt, hc + ctile * 16, kt * 32, TPAD, lane);
          c = wmma_bf16(af, bfr, c);
        }
        int cl = ctile * 16 + (lane & 15);
        // output-side channel permutation: r1=alp*4+u, r2=bet*4+v, r3=gam*2+w
        int gam = h % 3, bet = (h / 3) % 2, alp = h / 6;
        int w2 = cl & 1, v4 = (cl >> 1) & 3, u4 = cl >> 2;
        int pflat = ((alp * 4 + u4) * 8 + (bet * 4 + v4)) * 6 + (gam * 2 + w2);
#pragma unroll
        for (int r = 0; r < 8; ++r) {
          int stok = st * 16 + r + g8;
          if (stok < TOK) {
            int ho = m * 7 + stok / 7 + SHIFT; if (ho >= IMGD) ho -= IMGD;
            int wo = n * 7 + stok % 7 + SHIFT; if (wo >= IMGD) wo -= IMGD;
            out[(((size_t)b * IMGD + ho) * IMGD + wo) * EMB + pflat] = c[r];
          }
        }
      }
    }
  }
}

// ---------------------------------------------------------------------------
extern "C" void kernel_launch(void* const* d_in, const int* in_sizes, int n_in,
                              void* d_out, int out_size, void* d_ws,
                              size_t ws_size, hipStream_t stream) {
  (void)in_sizes; (void)n_in; (void)out_size; (void)ws_size;
  const float* x = (const float*)d_in[0];
  bf16*  Wd       = (bf16*)d_ws;                                   // 884736 B
  float* biasPerm = (float*)((char*)d_ws + (size_t)3 * EMB * EMB * sizeof(bf16));

  swmsa_prep<<<dim3(3 * EMB), dim3(EMB), 0, stream>>>(
      (const float*)d_in[1], (const float*)d_in[2], (const float*)d_in[3],
      (const float*)d_in[4], (const float*)d_in[5], (const float*)d_in[6],
      (const float*)d_in[7], (const float*)d_in[8], (const float*)d_in[9],
      (const float*)d_in[10], (const float*)d_in[11], (const float*)d_in[12],
      Wd, biasPerm);

  const size_t lds = (size_t)4 * TPAD * EMB * sizeof(bf16)   // X,Q,K,Vt
                   + (size_t)4 * 64 * 64 * sizeof(float)     // S
                   + (size_t)4 * 64 * 64 * sizeof(bf16);     // A^T  = 294912
  (void)hipFuncSetAttribute((const void*)swmsa_main,
                            hipFuncAttributeMaxDynamicSharedMemorySize,
                            (int)lds);
  swmsa_main<<<dim3(32 * 8 * 8), dim3(128), lds, stream>>>(x, Wd, biasPerm,
                                                           (float*)d_out);
}
